// MultiHeadAttention_61675730370767
// MI455X (gfx1250) — matrix operand order repaired and checked by reading, not compiled, not run on hardware
//
#include <hip/hip_runtime.h>
#include <hip/hip_bf16.h>
#include <math.h>

// ---------------------------------------------------------------------------
// MultiHeadAttention forward for MI455X (gfx1250), wave32 + WMMA f16.
// out  = (x @ Wo.T + bo)          [B,S,512]  f32   -> d_out[0 .. 2097151]
// attn = softmax(QK^T/sqrt(D))    [B,H,S,S]  f32   -> d_out[2097152 .. ]
// ---------------------------------------------------------------------------

#define HID 512
#define SEQ 2048
#define NB  2
#define NH  8
#define HD  64
#define SROW 2056                              // padded LDS row stride (floats)
#define SCORE_BYTES (32 * SROW * 4)            // 263168 B
#define VBUF_BYTES  (32 * HD * 2)              // 4096 B  (32x64 f16 V tile)
#define SMEM_BYTES  (SCORE_BYTES + VBUF_BYTES) // 267264 B < 320 KB WGP LDS

typedef __attribute__((ext_vector_type(16))) _Float16 v16h;
typedef __attribute__((ext_vector_type(8)))  float    v8f;

#define WMMA16(a, b, c) \
  __builtin_amdgcn_wmma_f32_16x16x32_f16(false, (a), false, (b), (short)0, (c), false, false)

// ---- fragment loaders (16x16x32 f16 WMMA) ---------------------------------
// A (16x32, MxK): lanes 0-15 -> M=0..15 K-base 0, lanes 16-31 -> K-base 16.
template <typename T>
__device__ __forceinline__ v16h ldA(const T* base, int ld, int lane) {
  const int m    = lane & 15;
  const int koff = (lane >> 4) << 4;
  const T* p = base + (size_t)m * ld + koff;
  v16h a;
#pragma unroll
  for (int v = 0; v < 8; ++v) {
    const int k = 2 * v;
    a[2 * v]     = (_Float16)p[k];
    a[2 * v + 1] = (_Float16)p[k + 1];
  }
  return a;
}

// B (32x16, KxN) from a source laid out (n, k): elem(n,k) = base[n*ld + k].
template <typename T>
__device__ __forceinline__ v16h ldBt(const T* base, int ld, int lane) {
  const int n  = lane & 15;
  const int k0 = (lane >> 4) << 4;
  const T* p = base + (size_t)n * ld + k0;
  v16h b;
#pragma unroll
  for (int v = 0; v < 8; ++v) {
    b[2 * v]     = (_Float16)p[2 * v];
    b[2 * v + 1] = (_Float16)p[2 * v + 1];
  }
  return b;
}

// B (32x16, KxN) from a row-major (k, n) source: elem(k,n) = base[k*ld + n].
template <typename T>
__device__ __forceinline__ v16h ldBrm(const T* base, int ld, int lane) {
  const int n  = lane & 15;
  const int k0 = (lane >> 4) << 4;
  v16h b;
#pragma unroll
  for (int v = 0; v < 8; ++v) {
    const int k = k0 + 2 * v;
    b[2 * v]     = (_Float16)base[(size_t)k * ld + n];
    b[2 * v + 1] = (_Float16)base[(size_t)(k + 1) * ld + n];
  }
  return b;
}

// ---- wave32 reductions via ds_swizzle (group-of-32 XOR) -------------------
template <int XORMASK>
__device__ __forceinline__ float swz_xor(float x) {
  return __int_as_float(
      __builtin_amdgcn_ds_swizzle(__float_as_int(x), ((XORMASK) << 10) | 0x1f));
}
__device__ __forceinline__ float wave_max32(float x) {
  x = fmaxf(x, swz_xor<16>(x));
  x = fmaxf(x, swz_xor<8>(x));
  x = fmaxf(x, swz_xor<4>(x));
  x = fmaxf(x, swz_xor<2>(x));
  x = fmaxf(x, swz_xor<1>(x));
  return x;
}
__device__ __forceinline__ float wave_sum32(float x) {
  x += swz_xor<16>(x);
  x += swz_xor<8>(x);
  x += swz_xor<4>(x);
  x += swz_xor<2>(x);
  x += swz_xor<1>(x);
  return x;
}

// ===========================================================================
// Kernel 1: fused Q/K/V projection.  [4096,512] x [512,512]^T + bias -> f16
// grid (32, 32, 3)  block 256 (8 waves); wave computes one 16x16 tile.
// ===========================================================================
__global__ void mha_qkv_proj(const float* __restrict__ q, const float* __restrict__ k,
                             const float* __restrict__ v,
                             const float* __restrict__ Wq, const float* __restrict__ bq,
                             const float* __restrict__ Wk, const float* __restrict__ bk,
                             const float* __restrict__ Wv, const float* __restrict__ bv,
                             _Float16* __restrict__ Qh, _Float16* __restrict__ Kh,
                             _Float16* __restrict__ Vh) {
  const int lane = threadIdx.x & 31;
  const int wave = threadIdx.x >> 5;
  const int m0 = (blockIdx.x * 8 + wave) * 16;
  const int n0 = blockIdx.y * 16;

  const float* src; const float* W; const float* bias; _Float16* dst;
  if (blockIdx.z == 0)      { src = q; W = Wq; bias = bq; dst = Qh; }
  else if (blockIdx.z == 1) { src = k; W = Wk; bias = bk; dst = Kh; }
  else                      { src = v; W = Wv; bias = bv; dst = Vh; }

  v8f c{};
#pragma unroll 4
  for (int kt = 0; kt < HID; kt += 32) {
    v16h a  = ldA (src + (size_t)m0 * HID + kt, HID, lane);
    v16h bb = ldBt(W   + (size_t)n0 * HID + kt, HID, lane);
    c = WMMA16(a, bb, c);
  }
  const int n  = lane & 15;
  const int mb = (lane >> 4) << 3;
  const float bval = bias[n0 + n];
  _Float16* o = dst + (size_t)m0 * HID + n0;
#pragma unroll
  for (int i = 0; i < 8; ++i)
    o[(size_t)(mb + i) * HID + n] = (_Float16)(c[i] + bval);
}

// ===========================================================================
// Kernel 2: attention.  One workgroup = (b,h) x 32 query rows.
// Phase 1: raw scores*scale (32x2048) -> LDS via WMMA.
// Phase 2: mask (int4 coalesced) + softmax in LDS, single b128 attn write.
// Phase 3: attn-strip @ V with V tile staged in LDS -> X (f16 ws).
// grid (16, 64)  block 256 (8 waves), 267 KB dynamic LDS.
// ===========================================================================
__global__ void mha_attention(const _Float16* __restrict__ Qh,
                              const _Float16* __restrict__ Kh,
                              const _Float16* __restrict__ Vh,
                              const int* __restrict__ mask,
                              float* __restrict__ attn_out,
                              _Float16* __restrict__ X) {
  extern __shared__ float smem[];                    // scores [32][SROW] + vbuf
  _Float16* vbuf = (_Float16*)(smem + 32 * SROW);    // [32][HD] f16 V tile
  const int lane = threadIdx.x & 31;
  const int wave = threadIdx.x >> 5;
  const int bh = blockIdx.x;
  const int b  = bh >> 3;
  const int h  = bh & 7;
  const int q0 = blockIdx.y * 32;
  const float scale = 0.125f;                        // 1/sqrt(64)
  const int n  = lane & 15;
  const int mb = (lane >> 4) << 3;

  // ---- Phase 1: scores = Q K^T * scale into LDS (mask applied in phase 2) --
  v16h aQ[2][2];
#pragma unroll
  for (int half = 0; half < 2; ++half)
#pragma unroll
    for (int kc = 0; kc < 2; ++kc)
      aQ[half][kc] = ldA(Qh + (size_t)(b * SEQ + q0 + half * 16) * HID + h * HD + kc * 32,
                         HID, lane);

  for (int t = wave; t < SEQ / 16; t += 8) {
    const int kcol0 = t * 16;
    const _Float16* kbase = Kh + (size_t)(b * SEQ + kcol0) * HID + h * HD;
    __builtin_prefetch(kbase + (size_t)8 * 16 * HID, 0, 0);   // next strip tile
    v16h b0 = ldBt(kbase,      HID, lane);
    v16h b1 = ldBt(kbase + 32, HID, lane);
#pragma unroll
    for (int half = 0; half < 2; ++half) {
      v8f c{};
      c = WMMA16(aQ[half][0], b0, c);
      c = WMMA16(aQ[half][1], b1, c);
#pragma unroll
      for (int i = 0; i < 8; ++i) {
        const int r = half * 16 + mb + i;            // 0..31 local row
        smem[(size_t)r * SROW + kcol0 + n] = c[i] * scale;
      }
    }
  }
  __syncthreads();

  // ---- Phase 2: mask + row softmax in LDS, single b128-streamed attn write -
#pragma unroll
  for (int j = 0; j < 4; ++j) {
    const int r = wave * 4 + j;
    float* row = smem + (size_t)r * SROW;
    const int* mrow = mask + (size_t)b * SEQ * SEQ + (size_t)(q0 + r) * SEQ;
    float m = -INFINITY;
    for (int c = lane * 4; c < SEQ; c += 128) {      // int4 mask + float4 LDS
      float4 sv = *(float4*)(row + c);
      const int4 mv = *(const int4*)(mrow + c);
      sv.x = (mv.x == 0) ? -1e10f : sv.x;
      sv.y = (mv.y == 0) ? -1e10f : sv.y;
      sv.z = (mv.z == 0) ? -1e10f : sv.z;
      sv.w = (mv.w == 0) ? -1e10f : sv.w;
      *(float4*)(row + c) = sv;
      m = fmaxf(m, fmaxf(fmaxf(sv.x, sv.y), fmaxf(sv.z, sv.w)));
    }
    m = wave_max32(m);
    float s = 0.f;
    for (int c = lane * 4; c < SEQ; c += 128) {
      float4 sv = *(float4*)(row + c);
      sv.x = __expf(sv.x - m); sv.y = __expf(sv.y - m);
      sv.z = __expf(sv.z - m); sv.w = __expf(sv.w - m);
      *(float4*)(row + c) = sv;
      s += sv.x + sv.y + sv.z + sv.w;
    }
    s = wave_sum32(s);
    const float inv = 1.0f / s;
    float* arow = attn_out + ((size_t)bh * SEQ + (q0 + r)) * SEQ;
    for (int c = lane * 4; c < SEQ; c += 128) {
      float4 sv = *(float4*)(row + c);
      sv.x *= inv; sv.y *= inv; sv.z *= inv; sv.w *= inv;
      *(float4*)(row + c) = sv;                      // keep for phase 3
      *(float4*)(arow + c) = sv;                     // 268 MB written exactly once
    }
  }
  __syncthreads();

  // ---- Phase 3: X = attn @ V, V tile staged through LDS ----
  const int half = wave >> 2;
  const int dt   = (wave & 3) * 16;
  const int vrow = threadIdx.x >> 3;                 // 0..31 (8 halves/thread)
  const int vcol = (threadIdx.x & 7) * 8;
  v8f c{};
  for (int kk = 0; kk < SEQ; kk += 32) {
    __syncthreads();                                 // vbuf reads of prev iter done
    // cooperative stage: 32x64 f16 V tile, one b128 per thread
    *(float4*)(vbuf + vrow * HD + vcol) =
        *(const float4*)(Vh + (size_t)(b * SEQ + kk + vrow) * HID + h * HD + vcol);
    __syncthreads();
    v16h a  = ldA  (smem + (size_t)(half * 16) * SROW + kk, SROW, lane);  // f32->f16
    v16h bb = ldBrm(vbuf + dt, HD, lane);
    c = WMMA16(a, bb, c);
  }
  _Float16* o = X + (size_t)(b * SEQ + q0 + half * 16) * HID + h * HD + dt;
#pragma unroll
  for (int i = 0; i < 8; ++i)
    o[(size_t)(mb + i) * HID + n] = (_Float16)c[i];
}

// ===========================================================================
// Kernel 3: output projection.  out = X @ Wo^T + bo  (f32 result)
// grid (32, 32)  block 256.
// ===========================================================================
__global__ void mha_out_proj(const _Float16* __restrict__ X,
                             const float* __restrict__ Wo, const float* __restrict__ bo,
                             float* __restrict__ out) {
  const int lane = threadIdx.x & 31;
  const int wave = threadIdx.x >> 5;
  const int m0 = (blockIdx.x * 8 + wave) * 16;
  const int n0 = blockIdx.y * 16;
  v8f c{};
#pragma unroll 4
  for (int kt = 0; kt < HID; kt += 32) {
    v16h a  = ldA (X  + (size_t)m0 * HID + kt, HID, lane);
    v16h bb = ldBt(Wo + (size_t)n0 * HID + kt, HID, lane);
    c = WMMA16(a, bb, c);
  }
  const int n  = lane & 15;
  const int mb = (lane >> 4) << 3;
  const float bval = bo[n0 + n];
  float* o = out + (size_t)m0 * HID + n0;
#pragma unroll
  for (int i = 0; i < 8; ++i)
    o[(size_t)(mb + i) * HID + n] = c[i] + bval;
}

// ===========================================================================
extern "C" void kernel_launch(void* const* d_in, const int* in_sizes, int n_in,
                              void* d_out, int out_size, void* d_ws, size_t ws_size,
                              hipStream_t stream) {
  const float* q   = (const float*)d_in[0];
  const float* k   = (const float*)d_in[1];
  const float* v   = (const float*)d_in[2];
  const int*  mask = (const int*)  d_in[3];
  const float* Wq  = (const float*)d_in[4];
  const float* bq  = (const float*)d_in[5];
  const float* Wk  = (const float*)d_in[6];
  const float* bk  = (const float*)d_in[7];
  const float* Wv  = (const float*)d_in[8];
  const float* bv  = (const float*)d_in[9];
  const float* Wo  = (const float*)d_in[10];
  const float* bo  = (const float*)d_in[11];

  const size_t T = (size_t)NB * SEQ * HID;           // 2,097,152 elements
  _Float16* Qh = (_Float16*)d_ws;
  _Float16* Kh = Qh + T;
  _Float16* Vh = Kh + T;
  _Float16* X  = Vh + T;                             // 16 MB total f16 ws

  float* out  = (float*)d_out;                       // [B,S,512]
  float* attn = out + T;                             // [B,H,S,S]

  (void)hipFuncSetAttribute((const void*)mha_attention,
                            hipFuncAttributeMaxDynamicSharedMemorySize, SMEM_BYTES);

  mha_qkv_proj<<<dim3(32, 32, 3), dim3(256), 0, stream>>>(
      q, k, v, Wq, bq, Wk, bk, Wv, bv, Qh, Kh, Vh);

  mha_attention<<<dim3(16, 64), dim3(256), SMEM_BYTES, stream>>>(
      Qh, Kh, Vh, mask, attn, X);

  mha_out_proj<<<dim3(32, 32), dim3(256), 0, stream>>>(X, Wo, bo, out);
}
